// MultiHeadAttention_84155589197934
// MI455X (gfx1250) — matrix operand order repaired and checked by reading, not compiled, not run on hardware
//
#include <hip/hip_runtime.h>

// ---------------------------------------------------------------------------
// Multi-head attention, MI455X (gfx1250).
//   B=2, L=4096, D=768, H=12, dk=64.  ~142 GFLOP.
//   All contractions: v_wmma_f32_16x16x32_bf16 (fp32 accum, fp32 softmax).
//   Shared operands staged via global_load_async_to_lds_b128 + double
//   buffering (ASYNCcnt + workgroup split barriers) -> 8x reuse per block.
//   Projection K-loops fully unrolled so accumulators stay in fixed VGPRs
//   (no loop-phi copies -> no WMMA->VALU hazard nops).
// ---------------------------------------------------------------------------

#define BATCH 2
#define SEQ   4096
#define DM    768
#define NH    12
#define DKH   64
#define ROWS  (BATCH * SEQ)   // 8192

typedef __attribute__((ext_vector_type(16))) __bf16 bf16x16;
typedef __attribute__((ext_vector_type(8)))  __bf16 bf16x8;
typedef __attribute__((ext_vector_type(8)))  float  f32x8;
typedef __attribute__((ext_vector_type(4)))  float  f32x4;
typedef __attribute__((ext_vector_type(4)))  unsigned short u16x4;

static __device__ inline unsigned short f2bf(float f) {
  unsigned u = __float_as_uint(f);
  u += 0x7FFFu + ((u >> 16) & 1u);
  return (unsigned short)(u >> 16);
}

static __device__ inline bf16x16 make16(bf16x8 lo, bf16x8 hi) {
  bf16x16 r;
#pragma unroll
  for (int i = 0; i < 8; ++i) { r[i] = lo[i]; r[i + 8] = hi[i]; }
  return r;
}

// A-tile (16x32 bf16): lane = half*16+lm holds row lm,
// K = k0 + half*16 + 0..15 (contiguous, matching the B-side LDS reads).
static __device__ inline bf16x16 load_a16x32(const unsigned short* rowptr,
                                             int k0, int half) {
  bf16x8 lo = *(const bf16x8*)(const void*)(rowptr + k0 + half * 16);
  bf16x8 hi = *(const bf16x8*)(const void*)(rowptr + k0 + half * 16 + 8);
  return make16(lo, hi);
}

static __device__ inline f32x8 wmma_bf16(bf16x16 a, bf16x16 b, f32x8 c) {
  return __builtin_amdgcn_wmma_f32_16x16x32_bf16(
      false, a, false, b, (short)0, c, false, false);
}

// Async DMA of one 16-byte chunk global -> LDS (tracked by ASYNCcnt).
static __device__ inline void async_copy16(const unsigned short* g,
                                           unsigned short* l) {
  unsigned lo = (unsigned)(size_t)(void*)l;         // LDS byte offset
  unsigned long long ga = (unsigned long long)(size_t)(const void*)g;
  asm volatile("global_load_async_to_lds_b128 %0, %1, off"
               :: "v"(lo), "v"(ga) : "memory");
}

static __device__ inline void wait_async0() {
  asm volatile("s_wait_asynccnt 0x0" ::: "memory");
}

// ---------------------------------------------------------------------------
// Kernel 1: fp32 -> bf16 convert (x4 vectorized; all sizes divisible by 4)
// ---------------------------------------------------------------------------
__global__ void cvt_f32_bf16(const float* __restrict__ src,
                             unsigned short* __restrict__ dst, int n4) {
  int i = blockIdx.x * blockDim.x + threadIdx.x;
  if (i < n4) {
    f32x4 v = *(const f32x4*)(const void*)(src + i * 4);
    u16x4 o;
#pragma unroll
    for (int e = 0; e < 4; ++e) o[e] = f2bf(v[e]);
    *(u16x4*)(void*)(dst + i * 4) = o;
  }
}

// ---------------------------------------------------------------------------
// Kernel 2: QKV projection, proj = x @ W^T + b.  Block = 8 waves sharing one
// 64-col weight strip; the 64x32 W-tile per k-step is async-staged to LDS
// (double buffered).  One wave = 32x64 output (2 m-tiles share each B-tile:
// 8 wmma per k-step).  K-loop fully unrolled -> accumulators never move.
//  z=0 -> Q (*0.125, [B,L,H*dk]); z=1 -> K ([B,L,H*dk]); z=2 -> V^T [B,H,dk,L]
// ---------------------------------------------------------------------------
__global__ __launch_bounds__(256) void qkv_proj_kernel(
    const unsigned short* __restrict__ xb,
    const unsigned short* __restrict__ Wqb,
    const unsigned short* __restrict__ Wkb,
    const unsigned short* __restrict__ Wvb,
    const float* __restrict__ bq, const float* __restrict__ bk,
    const float* __restrict__ bv,
    unsigned short* __restrict__ Qb, unsigned short* __restrict__ Kb,
    unsigned short* __restrict__ Vt) {
  __shared__ unsigned short wl[2 * 64 * 32];  // double-buffered 64n x 32k
  const int tid  = threadIdx.x;
  const int lane = tid & 31;
  const int wave = tid >> 5;
  const int half = lane >> 4;
  const int lm   = lane & 15;
  const int mu   = blockIdx.x * 8 + wave;  // 0..255 (32-row units)
  const int h    = blockIdx.y;             // 0..11
  const int z    = blockIdx.z;             // 0=Q 1=K 2=V
  const unsigned short* W   = (z == 0) ? Wqb : (z == 1) ? Wkb : Wvb;
  const float*          bia = (z == 0) ? bq : (z == 1) ? bk : bv;
  const int n0 = h * DKH;
  const int m0 = mu * 32;

  // cooperative async fill: thread -> one 16B chunk of the 64x32 tile
  const int wn = tid >> 2;        // 0..63
  const int wc = (tid & 3) * 8;   // 0,8,16,24
  const unsigned short* wsrc = W + (size_t)(n0 + wn) * DM + wc;

  f32x8 acc0[4], acc1[4];
#pragma unroll
  for (int t = 0; t < 4; ++t) {
    float bl = bia[n0 + t * 16 + lm];
#pragma unroll
    for (int e = 0; e < 8; ++e) { acc0[t][e] = bl; acc1[t][e] = bl; }
  }

  const unsigned short* arow0 = xb + (size_t)(m0 + lm) * DM;
  const unsigned short* arow1 = xb + (size_t)(m0 + 16 + lm) * DM;

  // prologue: stage k-step 0 into buffer 0
  async_copy16(wsrc, wl + wn * 32 + wc);

#pragma unroll
  for (int i = 0; i < DM / 32; ++i) {   // 24 fully-unrolled k-steps
    const int cur = i & 1;
    wait_async0();
    __syncthreads();                    // buffer `cur` ready for all waves
    if (i + 1 < DM / 32)
      async_copy16(wsrc + (i + 1) * 32,
                   wl + (cur ^ 1) * 2048 + wn * 32 + wc);

    bf16x16 a0 = load_a16x32(arow0, i * 32, half);
    bf16x16 a1 = load_a16x32(arow1, i * 32, half);
    const unsigned short* wt = wl + cur * 2048;
#pragma unroll
    for (int t = 0; t < 4; ++t) {
      bf16x16 b =
          *(const bf16x16*)(const void*)(wt + (t * 16 + lm) * 32 + half * 16);
      acc0[t] = wmma_bf16(a0, b, acc0[t]);
      acc1[t] = wmma_bf16(a1, b, acc1[t]);
    }
  }

  if (z == 2) {
#pragma unroll
    for (int t = 0; t < 4; ++t) {
      int d = t * 16 + lm;
#pragma unroll
      for (int r = 0; r < 8; ++r) {
#pragma unroll
        for (int s = 0; s < 2; ++s) {
          int mg  = m0 + s * 16 + r + 8 * half;
          int bb  = mg >> 12;
          int tok = mg & (SEQ - 1);
          float v = (s == 0) ? acc0[t][r] : acc1[t][r];
          Vt[((size_t)((bb * NH + h) * DKH + d)) * SEQ + tok] = f2bf(v);
        }
      }
    }
  } else {
    unsigned short* O = (z == 0) ? Qb : Kb;
    const float s = (z == 0) ? 0.125f : 1.0f;
#pragma unroll
    for (int t = 0; t < 4; ++t) {
#pragma unroll
      for (int r = 0; r < 8; ++r) {
        int mg0 = m0 + r + 8 * half;
        O[(size_t)mg0 * DM + n0 + t * 16 + lm] = f2bf(acc0[t][r] * s);
        O[(size_t)(mg0 + 16) * DM + n0 + t * 16 + lm] = f2bf(acc1[t][r] * s);
      }
    }
  }
}

// ---------------------------------------------------------------------------
// Kernel 3: flash attention.  Block = 8 waves (8 query tiles) of one (b,h);
// per 32-key block the shared K-tile (32x64) and V^T-tile (64x32) are
// async-staged into double-buffered LDS once and consumed by all waves.
// S = Q K^T (4 wmma), fp32 online softmax with half-wave shfl reductions,
// P restriped through per-wave LDS, O += P V (4 wmma).
// ---------------------------------------------------------------------------
__global__ __launch_bounds__(256) void flash_attn_kernel(
    const unsigned short* __restrict__ Qb,
    const unsigned short* __restrict__ Kb,
    const unsigned short* __restrict__ Vt,
    unsigned short* __restrict__ AO) {
  // [0,4096): K tiles (2 bufs x 32x64)  [4096,8192): V tiles (2 bufs x 64x32)
  // [8192,12288): per-wave P (8 x 16x32)
  __shared__ unsigned short sh[12288];
  const int tid  = threadIdx.x;
  const int lane = tid & 31;
  const int wave = tid >> 5;
  const int half = lane >> 4;
  const int lm   = lane & 15;
  unsigned short* plds = sh + 8192 + wave * 512;

  const int qt = blockIdx.x * 8 + wave;  // 0..255
  const int bh = blockIdx.y;             // 0..23
  const int b  = bh / NH;
  const int h  = bh % NH;
  const int q0 = qt * 16;

  const unsigned short* kbase = Kb + (size_t)b * SEQ * DM + h * DKH;
  const unsigned short* vbase = Vt + (size_t)((b * NH + h) * DKH) * SEQ;

  const int krow = tid >> 3, kc = (tid & 7) * 8;   // K: 32 rows x 128B
  const int vrow = tid >> 2, vc = (tid & 3) * 8;   // V: 64 rows x 64B

  const unsigned short* qrow = Qb + (size_t)(b * SEQ + q0 + lm) * DM + h * DKH;
  bf16x16 qa[2];
#pragma unroll
  for (int s = 0; s < 2; ++s) qa[s] = load_a16x32(qrow, s * 32, half);

  float mrow[8], lrow[8];
  f32x8 o[4];
#pragma unroll
  for (int r = 0; r < 8; ++r) { mrow[r] = -3.0e38f; lrow[r] = 0.0f; }
#pragma unroll
  for (int t = 0; t < 4; ++t)
#pragma unroll
    for (int e = 0; e < 8; ++e) o[t][e] = 0.0f;

  async_copy16(kbase + (size_t)krow * DM + kc, sh + krow * 64 + kc);
  async_copy16(vbase + (size_t)vrow * SEQ + vc, sh + 4096 + vrow * 32 + vc);

  const int NB = SEQ / 32;  // 128
  for (int i = 0; i < NB; ++i) {
    const int cur  = i & 1;
    const int kblk = i * 32;
    wait_async0();
    __syncthreads();  // buffer `cur` loaded; prior readers of `cur^1` done
    if (i + 1 < NB) {
      const int nb = kblk + 32;
      async_copy16(kbase + (size_t)(nb + krow) * DM + kc,
                   sh + (cur ^ 1) * 2048 + krow * 64 + kc);
      async_copy16(vbase + (size_t)vrow * SEQ + nb + vc,
                   sh + 4096 + (cur ^ 1) * 2048 + vrow * 32 + vc);
    }
    const unsigned short* kt = sh + cur * 2048;
    const unsigned short* vt = sh + 4096 + cur * 2048;

    f32x8 s0, s1;
#pragma unroll
    for (int e = 0; e < 8; ++e) { s0[e] = 0.0f; s1[e] = 0.0f; }
#pragma unroll
    for (int ds = 0; ds < 2; ++ds) {
      bf16x16 b0 = *(const bf16x16*)(const void*)(kt + lm * 64 +
                                                  ds * 32 + half * 16);
      bf16x16 b1 = *(const bf16x16*)(const void*)(kt + (16 + lm) * 64 +
                                                  ds * 32 + half * 16);
      s0 = wmma_bf16(qa[ds], b0, s0);
      s1 = wmma_bf16(qa[ds], b1, s1);
    }

    // online softmax; row m = r + 8*half lives in this lane's 16-lane half
#pragma unroll
    for (int r = 0; r < 8; ++r) {
      float v0 = s0[r], v1 = s1[r];
      float mx = fmaxf(v0, v1);
      mx = fmaxf(mx, __shfl_xor(mx, 1, 32));
      mx = fmaxf(mx, __shfl_xor(mx, 2, 32));
      mx = fmaxf(mx, __shfl_xor(mx, 4, 32));
      mx = fmaxf(mx, __shfl_xor(mx, 8, 32));
      float mo  = mrow[r];
      float mn  = fmaxf(mo, mx);
      float fac = __expf(mo - mn);
      float p0  = __expf(v0 - mn);
      float p1  = __expf(v1 - mn);
      float rs  = p0 + p1;
      rs += __shfl_xor(rs, 1, 32);
      rs += __shfl_xor(rs, 2, 32);
      rs += __shfl_xor(rs, 4, 32);
      rs += __shfl_xor(rs, 8, 32);
      lrow[r] = lrow[r] * fac + rs;
      mrow[r] = mn;
      o[0][r] *= fac; o[1][r] *= fac; o[2][r] *= fac; o[3][r] *= fac;
      plds[(r + 8 * half) * 32 + lm]      = f2bf(p0);
      plds[(r + 8 * half) * 32 + 16 + lm] = f2bf(p1);
    }
    asm volatile("s_wait_dscnt 0x0" ::: "memory");  // per-wave store->load

    bf16x8 plo = *(const bf16x8*)(const void*)(plds + lm * 32 + half * 16);
    bf16x8 phi = *(const bf16x8*)(const void*)(plds + lm * 32 + half * 16 + 8);
    bf16x16 pa = make16(plo, phi);

#pragma unroll
    for (int t = 0; t < 4; ++t) {
      bf16x16 vb =
          *(const bf16x16*)(const void*)(vt + (t * 16 + lm) * 32 + half * 16);
      o[t] = wmma_bf16(pa, vb, o[t]);
    }
  }

#pragma unroll
  for (int t = 0; t < 4; ++t) {
#pragma unroll
    for (int r = 0; r < 8; ++r) {
      int mg = b * SEQ + q0 + r + 8 * half;
      AO[(size_t)mg * DM + h * DKH + t * 16 + lm] = f2bf(o[t][r] / lrow[r]);
    }
  }
}

// ---------------------------------------------------------------------------
// Kernel 4: output projection, out = AO @ Wo^T + bo, fp32 result.
// Same scheme as QKV: LDS-staged weight strip, 32x64 per wave, full unroll.
// ---------------------------------------------------------------------------
__global__ __launch_bounds__(256) void out_proj_kernel(
    const unsigned short* __restrict__ AO,
    const unsigned short* __restrict__ Wob,
    const float* __restrict__ bo, float* __restrict__ out) {
  __shared__ unsigned short wl[2 * 64 * 32];
  const int tid  = threadIdx.x;
  const int lane = tid & 31;
  const int wave = tid >> 5;
  const int half = lane >> 4;
  const int lm   = lane & 15;
  const int mu   = blockIdx.x * 8 + wave;  // 0..255
  const int ns   = blockIdx.y;
  const int n0 = ns * 64;
  const int m0 = mu * 32;

  const int wn = tid >> 2;
  const int wc = (tid & 3) * 8;
  const unsigned short* wsrc = Wob + (size_t)(n0 + wn) * DM + wc;

  f32x8 acc0[4], acc1[4];
#pragma unroll
  for (int t = 0; t < 4; ++t) {
    float bl = bo[n0 + t * 16 + lm];
#pragma unroll
    for (int e = 0; e < 8; ++e) { acc0[t][e] = bl; acc1[t][e] = bl; }
  }

  const unsigned short* arow0 = AO + (size_t)(m0 + lm) * DM;
  const unsigned short* arow1 = AO + (size_t)(m0 + 16 + lm) * DM;

  async_copy16(wsrc, wl + wn * 32 + wc);

#pragma unroll
  for (int i = 0; i < DM / 32; ++i) {
    const int cur = i & 1;
    wait_async0();
    __syncthreads();
    if (i + 1 < DM / 32)
      async_copy16(wsrc + (i + 1) * 32,
                   wl + (cur ^ 1) * 2048 + wn * 32 + wc);

    bf16x16 a0 = load_a16x32(arow0, i * 32, half);
    bf16x16 a1 = load_a16x32(arow1, i * 32, half);
    const unsigned short* wt = wl + cur * 2048;
#pragma unroll
    for (int t = 0; t < 4; ++t) {
      bf16x16 b =
          *(const bf16x16*)(const void*)(wt + (t * 16 + lm) * 32 + half * 16);
      acc0[t] = wmma_bf16(a0, b, acc0[t]);
      acc1[t] = wmma_bf16(a1, b, acc1[t]);
    }
  }

#pragma unroll
  for (int t = 0; t < 4; ++t) {
#pragma unroll
    for (int r = 0; r < 8; ++r) {
      int mg0 = m0 + r + 8 * half;
      out[(size_t)mg0 * DM + n0 + t * 16 + lm] = acc0[t][r];
      out[(size_t)(mg0 + 16) * DM + n0 + t * 16 + lm] = acc1[t][r];
    }
  }
}

// ---------------------------------------------------------------------------
// Launch
// ---------------------------------------------------------------------------
extern "C" void kernel_launch(void* const* d_in, const int* in_sizes, int n_in,
                              void* d_out, int out_size, void* d_ws,
                              size_t ws_size, hipStream_t stream) {
  (void)in_sizes; (void)n_in; (void)out_size; (void)ws_size;

  const float* x  = (const float*)d_in[0];
  const float* Wq = (const float*)d_in[1];
  const float* bq = (const float*)d_in[2];
  const float* Wk = (const float*)d_in[3];
  const float* bk = (const float*)d_in[4];
  const float* Wv = (const float*)d_in[5];
  const float* bv = (const float*)d_in[6];
  const float* Wo = (const float*)d_in[7];
  const float* bo = (const float*)d_in[8];

  char* ws = (char*)d_ws;
  size_t off = 0;
  auto carve = [&](size_t bytes) -> unsigned short* {
    unsigned short* p = (unsigned short*)(ws + off);
    off += (bytes + 255) & ~(size_t)255;
    return p;
  };
  const size_t act_b = (size_t)ROWS * DM * 2;
  const size_t w_b   = (size_t)DM * DM * 2;
  unsigned short* xb  = carve(act_b);
  unsigned short* Wqb = carve(w_b);
  unsigned short* Wkb = carve(w_b);
  unsigned short* Wvb = carve(w_b);
  unsigned short* Wob = carve(w_b);
  unsigned short* Qb  = carve(act_b);
  unsigned short* Kb  = carve(act_b);
  unsigned short* Vt  = carve(act_b);
  unsigned short* AO  = carve(act_b);

  const int nx4 = ROWS * DM / 4;
  const int nw4 = DM * DM / 4;
  cvt_f32_bf16<<<(nx4 + 255) / 256, 256, 0, stream>>>(x, xb, nx4);
  cvt_f32_bf16<<<(nw4 + 255) / 256, 256, 0, stream>>>(Wq, Wqb, nw4);
  cvt_f32_bf16<<<(nw4 + 255) / 256, 256, 0, stream>>>(Wk, Wkb, nw4);
  cvt_f32_bf16<<<(nw4 + 255) / 256, 256, 0, stream>>>(Wv, Wvb, nw4);
  cvt_f32_bf16<<<(nw4 + 255) / 256, 256, 0, stream>>>(Wo, Wob, nw4);

  // 256 32-row units / 8 waves per block
  qkv_proj_kernel<<<dim3(32, 12, 3), 256, 0, stream>>>(
      xb, Wqb, Wkb, Wvb, bq, bk, bv, Qb, Kb, Vt);

  flash_attn_kernel<<<dim3(32, 24), 256, 0, stream>>>(Qb, Kb, Vt, AO);

  out_proj_kernel<<<dim3(32, 12), 256, 0, stream>>>(AO, Wob, bo,
                                                    (float*)d_out);
}
